// RealNVP2D_41154376630951
// MI455X (gfx1250) — matrix operand order repaired and checked by reading, not compiled, not run on hardware
//
#include <hip/hip_runtime.h>
#include <hip/hip_bf16.h>

#define NC      4
#define DD      2
#define H       512
#define TM      64          // batch rows per workgroup
#define LDH     520         // fp16 row stride in LDS (+8 pad -> bank spread)
#define THREADS 256

typedef __attribute__((ext_vector_type(16))) _Float16 v16h;
typedef __attribute__((ext_vector_type(8)))  _Float16 v8h;
typedef __attribute__((ext_vector_type(8)))  float    v8f;

union HFrag { v16h v; v8h h[2]; };
union HPack { v8h v; _Float16 e[8]; };

// A-matrix (16-bit, 16x32) lane layout: lane L, packed element e (=2*vgpr+half):
//   M = L%16,  K = (L/16)*8 + e + (e>=8 ? 8 : 0)
__device__ __host__ inline int a_klocal(int lane, int e) {
    return ((lane >> 4) * 8) + e + ((e >= 8) ? 8 : 0);
}

// ---------------------------------------------------------------------------
// Pack cw2 (NC,H,H) fp32 -> fp16 WMMA *A*-fragments of W^T:
//   A[m][k] = cw2[k][n0+m].  Tile = (kt: 32 K) x (mt: 16 output features).
//   dest[((i*512 + kt*32+mt)*32 + lane)*16 + e]
// ---------------------------------------------------------------------------
__global__ __launch_bounds__(THREADS) void pack_w2_kernel(
    const float* __restrict__ cw2, _Float16* __restrict__ w2p)
{
    int gid  = blockIdx.x * THREADS + threadIdx.x;   // 0 .. NC*H*H-1
    int i    = gid >> 18;
    int r    = gid & 262143;
    int tile = r >> 9;                               // kt*32 + mt
    int e512 = r & 511;
    int lane = e512 >> 4;
    int e    = e512 & 15;
    int kt   = tile >> 5;
    int mt   = tile & 31;
    int k    = kt * 32 + a_klocal(lane, e);
    int n    = mt * 16 + (lane & 15);
    w2p[gid] = (_Float16)cw2[((size_t)i * H + k) * H + n];
}

// ---------------------------------------------------------------------------
// Pack cw3 (NC,H,4) fp32 -> fp16 A-fragments of W3^T, padded to 16 rows.
// ---------------------------------------------------------------------------
__global__ __launch_bounds__(THREADS) void pack_w3_kernel(
    const float* __restrict__ cw3, _Float16* __restrict__ w3p)
{
    int gid  = blockIdx.x * THREADS + threadIdx.x;   // 0 .. NC*16*512-1
    int i    = gid >> 13;
    int r    = gid & 8191;
    int kt   = r >> 9;
    int e512 = r & 511;
    int lane = e512 >> 4;
    int e    = e512 & 15;
    int k    = kt * 32 + a_klocal(lane, e);
    int m    = lane & 15;
    float v  = (m < 4) ? cw3[((size_t)i * H + k) * 4 + m] : 0.0f;
    w3p[gid] = (_Float16)v;
}

// ---------------------------------------------------------------------------
// Fully fused RealNVP forward: all 4 couplings + final sigmoid/log-det.
// ---------------------------------------------------------------------------
__global__ __launch_bounds__(THREADS) void nvp_fused_kernel(
    const float* __restrict__ x,
    const float* __restrict__ cw1, const float* __restrict__ cb1,
    const float* __restrict__ cb2,
    const float* __restrict__ cb3,
    const float* __restrict__ an_logs, const float* __restrict__ an_b,
    const _Float16* __restrict__ w2p, const _Float16* __restrict__ w3p,
    float* __restrict__ out, float* __restrict__ logdet)
{
    extern __shared__ char smem[];
    _Float16* hA = (_Float16*)smem;                 // [TM][LDH]  layer-1 output
    _Float16* hB = hA + (size_t)TM * LDH;           // [TM][LDH]  layer-2 output
    float*    xb = (float*)(hB + (size_t)TM * LDH); // [TM][2]    running 'out'
    float*    ld = xb + TM * 2;                     // [TM]       running log_det
    float*    st = ld + TM;                         // [TM][4]    layer-3 result

    const int tid  = threadIdx.x;
    const int lane = tid & 31;
    const int wid  = tid >> 5;
    const int halfsel = lane >> 4;    // 0|1
    const int mrow    = lane & 15;
    const long rowG0 = (long)blockIdx.x * TM;

    if (tid < TM) {
        float2 v = ((const float2*)x)[rowG0 + tid];
        xb[tid * 2 + 0] = v.x;
        xb[tid * 2 + 1] = v.y;
        ld[tid] = 0.0f;
    }
    __syncthreads();

    for (int i = 0; i < NC; ++i) {
        const int j = i & 1;      // conditioning (kept) dim
        // ------------- layer 1: h1 = relu(x_j * w1_j + b1), vectorized x8 ---
        {
            const float* w1 = cw1 + ((size_t)i * DD + j) * H;
            const float* b1 = cb1 + (size_t)i * H;
            for (int c = tid; c < TM * (H / 8); c += THREADS) {
                int row = c >> 6;             // H/8 = 64 chunks per row
                int k8  = (c & 63) * 8;
                float xj = xb[row * 2 + j];
                HPack o;
                #pragma unroll
                for (int q = 0; q < 8; ++q)
                    o.e[q] = (_Float16)fmaf(xj, w1[k8 + q], b1[k8 + q]);
                v8h z = {};
                o.v = __builtin_elementwise_max(o.v, z);   // packed fp16 relu
                *(v8h*)(hA + (size_t)row * LDH + k8) = o.v;
            }
        }
        __syncthreads();

        // ------------- layer 2: h2 = relu(h1 @ cw2 + b2) via WMMA -----------
        // D = A x B, A = W^T tile (16 feat x 32 K), B = h1 (32 K x 16 rows).
        // One pass: each wave covers its full 64-feature slab (4 m-tiles) so
        // every h1 element is read from LDS exactly once per wave.
        {
            const _Float16* w2i = w2p + (size_t)i * H * H;
            const float*    b2  = cb2 + (size_t)i * H;
            const int mt0 = wid * 4;

            v8f acc[4][4];
            #pragma unroll
            for (int t = 0; t < 4; ++t) {
                // accumulators start at bias (8 consecutive features/lane)
                v8f bv = *(const v8f*)(b2 + (mt0 + t) * 16 + halfsel * 8);
                #pragma unroll
                for (int r = 0; r < 4; ++r) acc[t][r] = bv;
            }
            for (int kt = 0; kt < 16; ++kt) {
                // weight A-frags: 32 contiguous bytes per lane (global/L2)
                v16h aw[4];
                #pragma unroll
                for (int t = 0; t < 4; ++t) {
                    const _Float16* p = w2i +
                        (((size_t)(kt * 32 + mt0 + t)) << 9) + (lane << 4);
                    HFrag u;
                    u.h[0] = *(const v8h*)(p);
                    u.h[1] = *(const v8h*)(p + 8);
                    aw[t] = u.v;
                }
                // activation B-frags: two 16B chunks per lane, K = 8h+[0..7], 16+8h+[0..7] (LDS)
                v16h bx[4];
                #pragma unroll
                for (int r = 0; r < 4; ++r) {
                    const _Float16* p = hA +
                        (size_t)(r * 16 + mrow) * LDH + kt * 32 + halfsel * 8;
                    HFrag u;
                    u.h[0] = *(const v8h*)(p);
                    u.h[1] = *(const v8h*)(p + 16);
                    bx[r] = u.v;
                }
                #pragma unroll
                for (int t = 0; t < 4; ++t)
                    #pragma unroll
                    for (int r = 0; r < 4; ++r)
                        acc[t][r] = __builtin_amdgcn_wmma_f32_16x16x32_f16(
                            false, aw[t], false, bx[r],
                            (short)0, acc[t][r], false, false);
            }
            // fp16 convert, packed-half relu, ONE 16B LDS store per tile
            #pragma unroll
            for (int t = 0; t < 4; ++t) {
                #pragma unroll
                for (int r = 0; r < 4; ++r) {
                    HPack o;
                    #pragma unroll
                    for (int jj = 0; jj < 8; ++jj)
                        o.e[jj] = (_Float16)acc[t][r][jj];
                    v8h z = {};
                    o.v = __builtin_elementwise_max(o.v, z);
                    *(v8h*)(hB + (size_t)(r * 16 + mrow) * LDH
                               + (mt0 + t) * 16 + halfsel * 8) = o.v;
                }
            }
        }
        __syncthreads();

        // ------------- layer 3: st = h2 @ cw3 + b3 via WMMA (waves 0..3) ----
        if (wid < 4) {
            const _Float16* w3i = w3p + (size_t)i * 16 * 512;
            const int rowl = wid * 16 + mrow;
            v8f acc = (v8f){};
            for (int kt = 0; kt < 16; ++kt) {
                HFrag aw, bx;
                const _Float16* pa = w3i + (kt << 9) + (lane << 4);
                aw.h[0] = *(const v8h*)(pa);
                aw.h[1] = *(const v8h*)(pa + 8);
                const _Float16* pb = hB + (size_t)rowl * LDH + kt * 32 + halfsel * 8;
                bx.h[0] = *(const v8h*)(pb);
                bx.h[1] = *(const v8h*)(pb + 16);
                acc = __builtin_amdgcn_wmma_f32_16x16x32_f16(
                    false, aw.v, false, bx.v, (short)0, acc, false, false);
            }
            if (halfsel == 0) {           // lanes 0-15 hold c = 0..3 in acc[0..3]
                #pragma unroll
                for (int c = 0; c < 4; ++c)
                    st[rowl * 4 + c] = acc[c] + cb3[i * 4 + c];
            }
        }
        __syncthreads();

        // ------------- coupling epilogue + actnorm --------------------------
        if (tid < TM) {
            const int jp = 1 - j;
            float ls = tanhf(st[tid * 4 + jp]);
            float tt = st[tid * 4 + 2 + jp];
            float o0 = xb[tid * 2 + 0];
            float o1 = xb[tid * 2 + 1];
            if (jp == 0) o0 = o0 * __expf(ls) + tt;
            else         o1 = o1 * __expf(ls) + tt;
            float l = ld[tid] + ls;
            float al0 = an_logs[i * 2 + 0], al1 = an_logs[i * 2 + 1];
            o0 = o0 * __expf(al0) + an_b[i * 2 + 0];
            o1 = o1 * __expf(al1) + an_b[i * 2 + 1];
            l += al0 + al1;
            xb[tid * 2 + 0] = o0;
            xb[tid * 2 + 1] = o1;
            ld[tid] = l;
        }
        __syncthreads();
    }

    // ---- final: log_sigmoid(z)+log_sigmoid(-z) = -|z| - 2*log1p(e^-|z|) ----
    if (tid < TM) {
        float o0 = xb[tid * 2 + 0];
        float o1 = xb[tid * 2 + 1];
        float l  = ld[tid];
        float a0 = fabsf(o0), a1 = fabsf(o1);
        l += -a0 - 2.0f * log1pf(__expf(-a0));
        l += -a1 - 2.0f * log1pf(__expf(-a1));
        float s0 = 1.0f / (1.0f + __expf(-o0));
        float s1 = 1.0f / (1.0f + __expf(-o1));
        long row = rowG0 + tid;
        ((float2*)out)[row] = make_float2(s0, s1);
        logdet[row] = l;
    }
}

// ---------------------------------------------------------------------------
extern "C" void kernel_launch(void* const* d_in, const int* in_sizes, int n_in,
                              void* d_out, int out_size, void* d_ws, size_t ws_size,
                              hipStream_t stream)
{
    const float* x       = (const float*)d_in[0];
    const float* cw1     = (const float*)d_in[1];
    const float* cb1     = (const float*)d_in[2];
    const float* cw2     = (const float*)d_in[3];
    const float* cb2     = (const float*)d_in[4];
    const float* cw3     = (const float*)d_in[5];
    const float* cb3     = (const float*)d_in[6];
    const float* an_logs = (const float*)d_in[7];
    const float* an_b    = (const float*)d_in[8];

    const int B = in_sizes[0] / 2;
    float* out    = (float*)d_out;
    float* logdet = out + (size_t)B * 2;
    _Float16* w2p = (_Float16*)d_ws;                       // 2 MB packed
    _Float16* w3p = w2p + (size_t)NC * H * H;              // 64 KB packed

    pack_w2_kernel<<<(NC * H * H) / THREADS, THREADS, 0, stream>>>(cw2, w2p);
    pack_w3_kernel<<<(NC * 16 * 512) / THREADS, THREADS, 0, stream>>>(cw3, w3p);

    size_t smem = (size_t)2 * TM * LDH * sizeof(_Float16)
                + (size_t)(TM * 2 + TM + TM * 4) * sizeof(float);
    nvp_fused_kernel<<<B / TM, THREADS, smem, stream>>>(
        x, cw1, cb1, cb2, cb3, an_logs, an_b, w2p, w3p, out, logdet);
}
